// SimpleLSTMCell_16243566313496
// MI455X (gfx1250) — matrix operand order repaired and checked by reading, not compiled, not run on hardware
//
#include <hip/hip_runtime.h>

// ---------------------------------------------------------------------------
// LSTM cell on MI455X (gfx1250): 4-gate fused GEMM via v_wmma_f32_16x16x32_f16
//   gates = x @ W_g^T + h @ U_g^T + b  (g = i,f,o,c), then elementwise epilogue
//
// Per-wave tile: M=64 (4 WMMA tiles) x J=16 x 4 gates -> 16 acc tiles,
// 16 WMMAs per K=32 step against 16 b128 loads (all base+imm addressed).
// ---------------------------------------------------------------------------

#define BATCH 16384
#define DIM   1024
#define HID   1024

typedef __attribute__((ext_vector_type(16))) _Float16 v16h;
typedef __attribute__((ext_vector_type(8)))  _Float16 v8h;
typedef __attribute__((ext_vector_type(4)))  _Float16 v4h;
typedef __attribute__((ext_vector_type(8)))  float    v8f;

union HFrag { v16h v; v8h h[2]; };

// ---------------------------------------------------------------------------
// f32 -> f16 conversion (4-wide)
// ---------------------------------------------------------------------------
__global__ void cvt_f32_to_f16(const float* __restrict__ src,
                               _Float16* __restrict__ dst, int n4) {
  int i = blockIdx.x * blockDim.x + threadIdx.x;
  if (i < n4) {
    float4 vv = ((const float4*)src)[i];
    v4h o = { (_Float16)vv.x, (_Float16)vv.y, (_Float16)vv.z, (_Float16)vv.w };
    ((v4h*)dst)[i] = o;
  }
}

// ---------------------------------------------------------------------------
// Fragment assembly from two preloaded base pointers + k offset.
// A 16x32 (MxK) wave32 layout: lanes 0-15 row M=lane&15, K = {0..7, 16..23};
// lanes 16-31 same rows, K = {8..15, 24..31}. Per lane: 16B at kb, 16B at kb+32B.
// B 32x16 (KxN) wave32 layout: lanes 0-15 col N=lane&15, K = {0..7, 16..23};
// lanes 16-31 K = {8..15, 24..31}. Per lane: 16B at kb, 16B at kb+32B.
// ---------------------------------------------------------------------------
__device__ __forceinline__ v16h frag_a(const _Float16* __restrict__ p, int k0) {
  HFrag f;
  f.h[0] = *(const v8h*)(p + k0);
  f.h[1] = *(const v8h*)(p + k0 + 16);
  return f.v;
}
__device__ __forceinline__ v16h frag_b(const _Float16* __restrict__ p, int k0) {
  HFrag f;
  f.h[0] = *(const v8h*)(p + k0);
  f.h[1] = *(const v8h*)(p + k0 + 16);
  return f.v;
}

__device__ __forceinline__ float sig_fast(float x) {
  return 1.0f / (1.0f + __expf(-x));
}
__device__ __forceinline__ float tanh_fast(float x) {
  float t = __expf(-2.0f * fabsf(x));         // in (0,1], no overflow
  float r = (1.0f - t) / (1.0f + t);
  return copysignf(r, x);
}

// ---------------------------------------------------------------------------
// Fused 4-gate GEMM + LSTM epilogue.
// Block = 256 threads (8 waves): 2 waves along M x 4 waves along J.
// Grid: (HID/64, BATCH/128).
// ---------------------------------------------------------------------------
__global__ __launch_bounds__(256) void lstm_wmma_kernel(
    const _Float16* __restrict__ xh, const _Float16* __restrict__ hh,
    const _Float16* __restrict__ Wi, const _Float16* __restrict__ Wf,
    const _Float16* __restrict__ Wo, const _Float16* __restrict__ Wc,
    const _Float16* __restrict__ Ui, const _Float16* __restrict__ Uf,
    const _Float16* __restrict__ Uo, const _Float16* __restrict__ Uc,
    const float* __restrict__ bWi, const float* __restrict__ bWf,
    const float* __restrict__ bWo, const float* __restrict__ bWc,
    const float* __restrict__ bUi, const float* __restrict__ bUf,
    const float* __restrict__ bUo, const float* __restrict__ bUc,
    const float* __restrict__ cin,
    float* __restrict__ hout, float* __restrict__ cout_)
{
  const int lane = threadIdx.x & 31;
  const int w    = threadIdx.x >> 5;
  const int wm   = w & 1;                     // 0..1 along M
  const int wj   = w >> 1;                    // 0..3 along J
  const int m0   = blockIdx.y * 128 + wm * 64;
  const int j0   = blockIdx.x * 64  + wj * 16;

  const int l15 = lane & 15;
  const int kbA = (lane >> 4) << 3;           // 0 or 8
  const int kbB = (lane >> 4) << 3;           // 0 or 8

  v8f acc[4][4];                              // [gate][m-subtile]
#pragma unroll
  for (int g = 0; g < 4; ++g)
#pragma unroll
    for (int t = 0; t < 4; ++t)
      acc[g][t] = (v8f){0.f, 0.f, 0.f, 0.f, 0.f, 0.f, 0.f, 0.f};

  const _Float16* Asrc[2] = {xh, hh};
  const _Float16* Bsrc[2][4] = {{Wi, Wf, Wo, Wc}, {Ui, Uf, Uo, Uc}};

  for (int ph = 0; ph < 2; ++ph) {
    // Fixed per-lane base pointers: inner loop addresses are base + imm.
    const _Float16* pA[4];
#pragma unroll
    for (int t = 0; t < 4; ++t)
      pA[t] = Asrc[ph] + (size_t)(m0 + t * 16 + l15) * DIM + kbA;
    const _Float16* pB[4];
#pragma unroll
    for (int g = 0; g < 4; ++g)
      pB[g] = Bsrc[ph][g] + (size_t)(j0 + l15) * DIM + kbB;

#pragma unroll 4
    for (int k0 = 0; k0 < DIM; k0 += 32) {
      v16h a[4], b[4];
#pragma unroll
      for (int t = 0; t < 4; ++t) a[t] = frag_a(pA[t], k0);
#pragma unroll
      for (int g = 0; g < 4; ++g) b[g] = frag_b(pB[g], k0);
#pragma unroll
      for (int g = 0; g < 4; ++g)
#pragma unroll
        for (int t = 0; t < 4; ++t)
          acc[g][t] = __builtin_amdgcn_wmma_f32_16x16x32_f16(
              false, a[t], false, b[g], (short)0, acc[g][t], false, false);
    }
  }

  // ---- Epilogue ------------------------------------------------------------
  // C/D layout: VGPR r, lanes 0-15 -> M=r, N=lane; lanes 16-31 -> M=r+8.
  const int col = j0 + l15;
  const float bi = bWi[col] + bUi[col];
  const float bf = bWf[col] + bUf[col];
  const float bo = bWo[col] + bUo[col];
  const float bc = bWc[col] + bUc[col];
  const int rbase = m0 + ((lane >> 4) << 3);

#pragma unroll
  for (int tm = 0; tm < 4; ++tm) {
#pragma unroll
    for (int r = 0; r < 8; ++r) {
      const int row = rbase + tm * 16 + r;
      const size_t idx = (size_t)row * HID + col;
      float iv = sig_fast(acc[0][tm][r] + bi);
      float fv = sig_fast(acc[1][tm][r] + bf);
      float ov = sig_fast(acc[2][tm][r] + bo);
      float ct = tanh_fast(acc[3][tm][r] + bc);
      float cn = fv * cin[idx] + iv * ct;
      cout_[idx] = cn;
      hout[idx]  = ov * tanh_fast(cn);
    }
  }
}

// ---------------------------------------------------------------------------
// Host launcher
// Input order: 0:x 1:h 2:c 3:W_i 4:b_Wi 5:U_i 6:b_Ui 7:W_f 8:b_Wf 9:U_f
//              10:b_Uf 11:W_o 12:b_Wo 13:U_o 14:b_Uo 15:W_c 16:b_Wc 17:U_c 18:b_Uc
// Output: h_next [B,H] then c_next [B,H], f32.
// ---------------------------------------------------------------------------
extern "C" void kernel_launch(void* const* d_in, const int* in_sizes, int n_in,
                              void* d_out, int out_size, void* d_ws, size_t ws_size,
                              hipStream_t stream) {
  (void)in_sizes; (void)n_in; (void)out_size; (void)ws_size;

  const float* x = (const float*)d_in[0];
  const float* h = (const float*)d_in[1];
  const float* c = (const float*)d_in[2];
  const float* W[4]  = {(const float*)d_in[3],  (const float*)d_in[7],
                        (const float*)d_in[11], (const float*)d_in[15]};
  const float* bW[4] = {(const float*)d_in[4],  (const float*)d_in[8],
                        (const float*)d_in[12], (const float*)d_in[16]};
  const float* U[4]  = {(const float*)d_in[5],  (const float*)d_in[9],
                        (const float*)d_in[13], (const float*)d_in[17]};
  const float* bU[4] = {(const float*)d_in[6],  (const float*)d_in[10],
                        (const float*)d_in[14], (const float*)d_in[18]};

  // Workspace layout (f16): x[16M] h[16M] W_i..W_c[1M each] U_i..U_c[1M each]
  char* ws = (char*)d_ws;
  _Float16* xh = (_Float16*)ws;
  _Float16* hh = (_Float16*)(ws + (size_t)BATCH * DIM * sizeof(_Float16));
  size_t off = (size_t)2 * BATCH * DIM * sizeof(_Float16);
  _Float16 *Wh[4], *Uh[4];
  for (int g = 0; g < 4; ++g) {
    Wh[g] = (_Float16*)(ws + off);
    off += (size_t)HID * DIM * sizeof(_Float16);
  }
  for (int g = 0; g < 4; ++g) {
    Uh[g] = (_Float16*)(ws + off);
    off += (size_t)HID * HID * sizeof(_Float16);
  }

  auto cvt = [&](const float* s, _Float16* d, size_t n) {
    int n4 = (int)(n / 4);
    cvt_f32_to_f16<<<(n4 + 255) / 256, 256, 0, stream>>>(s, d, n4);
  };
  cvt(x, xh, (size_t)BATCH * DIM);
  cvt(h, hh, (size_t)BATCH * HID);
  for (int g = 0; g < 4; ++g) cvt(W[g], Wh[g], (size_t)HID * DIM);
  for (int g = 0; g < 4; ++g) cvt(U[g], Uh[g], (size_t)HID * HID);

  float* hout  = (float*)d_out;
  float* cout_ = hout + (size_t)BATCH * HID;

  dim3 grid(HID / 64, BATCH / 128);
  lstm_wmma_kernel<<<grid, 256, 0, stream>>>(
      xh, hh,
      Wh[0], Wh[1], Wh[2], Wh[3],
      Uh[0], Uh[1], Uh[2], Uh[3],
      bW[0], bW[1], bW[2], bW[3],
      bU[0], bU[1], bU[2], bU[3],
      c, hout, cout_);
}
